// PointNet2_46282567582125
// MI455X (gfx1250) — compile-verified, matrix operand order checked by reading, not run on hardware
//
#include <hip/hip_runtime.h>

typedef _Float16 half_t;
typedef __attribute__((ext_vector_type(16))) _Float16 v16h;
typedef __attribute__((ext_vector_type(8)))  _Float16 v8h;
typedef __attribute__((ext_vector_type(8)))  float    v8f;

// ---------------- problem constants ----------------
constexpr int B_  = 8;
constexpr int N1  = 4096;
constexpr int M1  = 2048;
constexpr int M2  = 512;
constexpr int KNB = 64;
constexpr int E1  = B_ * M1 * KNB;   // 1,048,576 edges (SA1)
constexpr int E2  = B_ * M2 * KNB;   // 262,144 edges (SA2)
constexpr int R3  = B_ * M2;         // 4096 rows (mlp3)

// ---------------- workspace layout (bytes) ----------------
constexpr size_t O_POS   = 0;                              // f32 [8,4096,3]
constexpr size_t O_IDX1  = 1ull << 19;                     // i32 [8,2048]
constexpr size_t O_CENT1 = O_IDX1  + (1ull << 17);         // f32 [8,2048,3]
constexpr size_t O_NBR1  = O_CENT1 + (1ull << 18);         // i32 [8,2048,64]
constexpr size_t O_VAL1  = O_NBR1  + (4ull << 20);         // i32 [8,2048,64]
constexpr size_t O_IDX2  = O_VAL1  + (4ull << 20);         // i32 [8,512]
constexpr size_t O_CENT2 = O_IDX2  + (1ull << 16);         // f32 [8,512,3]
constexpr size_t O_NBR2  = O_CENT2 + (1ull << 16);         // i32 [8,512,64]
constexpr size_t O_VAL2  = O_NBR2  + (1ull << 20);         // i32 [8,512,64]
constexpr size_t O_H1    = O_VAL2  + (1ull << 20);         // f16 [8,2048,128]
constexpr size_t O_H2    = O_H1    + (4ull << 20);         // f16 [8,512,256]
constexpr size_t O_G     = O_H2    + (2ull << 20);         // f32 [8,1024]
constexpr size_t O_TMP   = O_G     + (1ull << 16);         // f32 [8,256]
constexpr size_t O_STATS = O_TMP   + (1ull << 16);         // f32 [2049]: sums, sumsq, cnt
constexpr size_t O_WH    = O_STATS + (1ull << 16);         // f16 packed weights, ~1.7MB
constexpr size_t O_BUF0  = 32ull << 20;                    // f16 ping (256MB)
constexpr size_t O_BUF1  = O_BUF0 + (272ull << 20);        // f16 pong (256MB)

// fragment half-index -> k offset within a 32-k block, given lane half kb (0 or 8)
__device__ __host__ inline int frag_k(int j) {
    int p = j >> 1;
    int kk = (p < 4) ? (2 * p) : (16 + 2 * (p - 4));
    return kk + (j & 1);
}

// ---------------- small utility kernels ----------------
__global__ void zero_kernel(float* p, int n) {
    int i = blockIdx.x * blockDim.x + threadIdx.x;
    if (i < n) p[i] = 0.0f;
}
__global__ void set_scalar(float* p, float v) {
    if (blockIdx.x == 0 && threadIdx.x == 0) p[0] = v;
}
__global__ void count_valid(const int* __restrict__ valid, float* cnt, int n) {
    __shared__ int s[256];
    int t = threadIdx.x;
    int v = 0;
    for (int i = blockIdx.x * blockDim.x + t; i < n; i += gridDim.x * blockDim.x)
        v += valid[i];
    s[t] = v; __syncthreads();
    for (int off = 128; off > 0; off >>= 1) {
        if (t < off) s[t] += s[t + off];
        __syncthreads();
    }
    if (t == 0) atomicAdd(cnt, (float)s[0]);
}

// pos[b,i,c] = data[b,c,i]
__global__ void transpose_pos(const float* __restrict__ data, float* __restrict__ pos, int Npts) {
    int i = blockIdx.x * blockDim.x + threadIdx.x;       // B*Npts*3
    if (i >= B_ * Npts * 3) return;
    int c = i % 3, bi = i / 3;
    int b = bi / Npts, p = bi % Npts;
    pos[i] = data[((size_t)b * 3 + c) * Npts + p];
}

// ---------------- farthest point sampling (one WG per batch) ----------------
__global__ void fps_kernel(const float* __restrict__ pos, int* __restrict__ idxOut,
                           int Npts, int M) {
    __shared__ float sd[4096];
    __shared__ float rv[256];
    __shared__ int   ri[256];
    __shared__ int   curIdx;
    __shared__ float cx, cy, cz;
    int b = blockIdx.x, t = threadIdx.x;
    const float* P = pos + (size_t)b * Npts * 3;
    int* out = idxOut + (size_t)b * M;
    float x0 = P[0], y0 = P[1], z0 = P[2];
    for (int i = t; i < Npts; i += 256) {
        float dx = P[i*3] - x0, dy = P[i*3+1] - y0, dz = P[i*3+2] - z0;
        sd[i] = dx*dx + dy*dy + dz*dz;
    }
    if (t == 0) out[0] = 0;
    __syncthreads();
    for (int s = 1; s < M; ++s) {
        float bv = -1.0f; int bi = 0x7fffffff;
        for (int i = t; i < Npts; i += 256) {
            float v = sd[i];
            if (v > bv) { bv = v; bi = i; }   // ascending i keeps first max
        }
        rv[t] = bv; ri[t] = bi; __syncthreads();
        for (int off = 128; off > 0; off >>= 1) {
            if (t < off) {
                if (rv[t+off] > rv[t] || (rv[t+off] == rv[t] && ri[t+off] < ri[t])) {
                    rv[t] = rv[t+off]; ri[t] = ri[t+off];
                }
            }
            __syncthreads();
        }
        if (t == 0) {
            int sel = ri[0];
            out[s] = sel; curIdx = sel;
            cx = P[sel*3]; cy = P[sel*3+1]; cz = P[sel*3+2];
        }
        __syncthreads();
        float px = cx, py = cy, pz = cz;
        for (int i = t; i < Npts; i += 256) {
            float dx = P[i*3] - px, dy = P[i*3+1] - py, dz = P[i*3+2] - pz;
            float nd = dx*dx + dy*dy + dz*dz;
            if (nd < sd[i]) sd[i] = nd;
        }
        __syncthreads();
    }
}

__global__ void gather_centers(const float* __restrict__ pos, const int* __restrict__ idx,
                               float* __restrict__ centers, int Npts, int M) {
    int i = blockIdx.x * blockDim.x + threadIdx.x;  // B*M*3
    if (i >= B_ * M * 3) return;
    int c = i % 3, bm = i / 3;
    int b = bm / M;
    centers[i] = pos[((size_t)b * Npts + idx[bm]) * 3 + c];
}

// ---------------- ball query: K nearest within radius (one WG per center) -------
__global__ void ball_query_kernel(const float* __restrict__ pos, const float* __restrict__ centers,
                                  int* __restrict__ nbr, int* __restrict__ valid,
                                  int Npts, int Mc, float r2, int K) {
    __shared__ float key[4096];
    __shared__ float rv[64];
    __shared__ int   ri[64];
    int cidx = blockIdx.x;              // b*Mc + m
    int b = cidx / Mc;
    int t = threadIdx.x;
    const float* P = pos + (size_t)b * Npts * 3;
    float cx = centers[(size_t)cidx*3], cy = centers[(size_t)cidx*3+1], cz = centers[(size_t)cidx*3+2];
    for (int i = t; i < Npts; i += 64) {
        float dx = P[i*3] - cx, dy = P[i*3+1] - cy, dz = P[i*3+2] - cz;
        float d2 = dx*dx + dy*dy + dz*dz;
        key[i] = (d2 <= r2) ? d2 : 1.0e30f;   // out-of-radius sentinel
    }
    __syncthreads();
    for (int kk = 0; kk < K; ++kk) {
        float bv = 3.0e38f; int bi = 0x7fffffff;
        for (int i = t; i < Npts; i += 64) {
            float v = key[i];
            if (v < bv) { bv = v; bi = i; }   // ascending keeps lowest idx on ties
        }
        rv[t] = bv; ri[t] = bi; __syncthreads();
        for (int off = 32; off > 0; off >>= 1) {
            if (t < off) {
                if (rv[t+off] < rv[t] || (rv[t+off] == rv[t] && ri[t+off] < ri[t])) {
                    rv[t] = rv[t+off]; ri[t] = ri[t+off];
                }
            }
            __syncthreads();
        }
        if (t == 0) {
            int sel = ri[0];
            nbr[(size_t)cidx * K + kk]   = sel;
            valid[(size_t)cidx * K + kk] = (rv[0] < 1.0e30f) ? 1 : 0;
            key[sel] = 2.0e38f;               // taken sentinel
        }
        __syncthreads();
    }
}

// ---------------- edge feature builders (f16, K-padded) ----------------
__global__ void build_edges1(const float* __restrict__ pos, const float* __restrict__ centers,
                             const int* __restrict__ nbr, half_t* __restrict__ X,
                             int Npts, int Mc) {
    long long i = (long long)blockIdx.x * blockDim.x + threadIdx.x;  // E1*32
    if (i >= (long long)E1 * 32) return;
    int j = (int)(i & 31);
    int e = (int)(i >> 5);
    half_t v = (half_t)0.0f;
    if (j < 3) {
        int cidx = e >> 6;
        int b = cidx / Mc;
        int p = nbr[e];
        v = (half_t)(pos[((size_t)b * Npts + p) * 3 + j] - centers[(size_t)cidx * 3 + j]);
    }
    X[(size_t)e * 32 + j] = v;
}

__global__ void build_edges2(const float* __restrict__ pos, const float* __restrict__ centers,
                             const int* __restrict__ nbr, const half_t* __restrict__ H1,
                             half_t* __restrict__ X, int Npts, int Mc) {
    long long i = (long long)blockIdx.x * blockDim.x + threadIdx.x;  // E2*160
    if (i >= (long long)E2 * 160) return;
    int j = (int)(i % 160);
    int e = (int)(i / 160);
    int cidx = e >> 6;
    int b = cidx / Mc;
    half_t v = (half_t)0.0f;
    if (j < 128) {
        int p = nbr[e];
        v = H1[((size_t)b * Npts + p) * 128 + j];
    } else if (j < 131) {
        int p = nbr[e];
        int c = j - 128;
        v = (half_t)(pos[((size_t)b * Npts + p) * 3 + c] - centers[(size_t)cidx * 3 + c]);
    }
    X[(size_t)e * 160 + j] = v;
}

__global__ void build_x3(const half_t* __restrict__ h2, const float* __restrict__ cent2,
                         half_t* __restrict__ X) {
    int i = blockIdx.x * blockDim.x + threadIdx.x;  // R3*288
    if (i >= R3 * 288) return;
    int r = i / 288, j = i % 288;
    half_t v;
    if (j < 256)       v = h2[(size_t)r * 256 + j];
    else if (j < 259)  v = (half_t)cent2[r * 3 + (j - 256)];
    else               v = (half_t)0.0f;
    X[(size_t)r * 288 + j] = v;
}

// ---------------- weight repack: f32 W[Kin,Cout] -> f16 WMMA-fragment order ------
// Packed layout: [(colTile ct, kBlock kblk, lane L)] -> 16 contiguous halfs.
// Lane L covers col = ct*16 + (L&15), k-half kb = (L<16)?0:8.
// Element j holds W[kblk*32 + kb + frag_k(j)][col], zero-padded past Kin.
__global__ void pack_w(const float* __restrict__ W, half_t* __restrict__ Wp,
                       int Kin, int Kpad, int Cout) {
    int i = blockIdx.x * blockDim.x + threadIdx.x;   // Kpad*Cout packed elements
    if (i >= Kpad * Cout) return;
    int j    = i & 15;
    int lane = (i >> 4) & 31;
    int rest = i >> 9;                                // ct*numKB + kblk
    int numKB = Kpad >> 5;
    int kblk = rest % numKB;
    int ct   = rest / numKB;
    int col = ct * 16 + (lane & 15);
    int kb  = (lane < 16) ? 0 : 8;
    int k   = kblk * 32 + kb + frag_k(j);
    Wp[i] = (k < Kin) ? (half_t)W[(size_t)k * Cout + col] : (half_t)0.0f;
}

// ---------------- WMMA GEMM: Y = relu(X @ W + bias), f16 in / f32 acc / f16 out ----
// One wave computes a 16-row x 64-col strip (4 WMMA 16x16x32 tiles, A-frag reused).
// A fragment: two contiguous 16B loads per lane. B fragment: one contiguous 32B
// load per lane from the pre-packed weight buffer (coalesced across lanes).
__global__ void gemm_bias_relu(const half_t* __restrict__ X, const half_t* __restrict__ Wp,
                               const float* __restrict__ bias, half_t* __restrict__ Y,
                               int rows, int Kdim, int Cout) {
    const int lane = threadIdx.x & 31;
    const int wave = threadIdx.x >> 5;
    const int wavesPerBlock = blockDim.x >> 5;
    long long tile = (long long)blockIdx.x * wavesPerBlock + wave;
    const int colGroups = Cout >> 6;                 // 64 cols per wave
    const long long rowTiles = rows >> 4;
    if (tile >= rowTiles * colGroups) return;
    const int rt = (int)(tile / colGroups);
    const int cg = (int)(tile % colGroups);

    const int row = rt * 16 + (lane & 15);
    const int kb  = (lane < 16) ? 0 : 8;             // A/B fragment K-half select
    const int numKB = Kdim >> 5;

    v8f acc[4] = {v8f{}, v8f{}, v8f{}, v8f{}};

    const half_t* xrow = X + (size_t)row * Kdim + kb;

    for (int kblk = 0; kblk < numKB; ++kblk) {
        // A fragment: halfs k = kb+0..7 and kb+16..23 of this 32-k block
        const half_t* xr = xrow + kblk * 32;
        v8h alo = *(const v8h*)(xr);
        v8h ahi = *(const v8h*)(xr + 16);
        v16h a = __builtin_shufflevector(alo, ahi,
                                         0, 1, 2, 3, 4, 5, 6, 7,
                                         8, 9, 10, 11, 12, 13, 14, 15);
        #pragma unroll
        for (int c = 0; c < 4; ++c) {
            int ct = cg * 4 + c;
            const v16h* bp = (const v16h*)(Wp + ((((size_t)ct * numKB + kblk) * 32 + lane) << 4));
            v16h bf = *bp;
            acc[c] = __builtin_amdgcn_wmma_f32_16x16x32_f16(
                false, a, false, bf, (short)0, acc[c], false, false);
        }
    }

    // epilogue: bias + relu, store f16. D layout: col = lane&15, row = i + 8*(lane>=16)
    const int rbase = rt * 16 + ((lane >> 4) << 3);
    #pragma unroll
    for (int c = 0; c < 4; ++c) {
        int col = cg * 64 + c * 16 + (lane & 15);
        float bv = bias[col];
        #pragma unroll
        for (int i = 0; i < 8; ++i) {
            int r = rbase + i;
            float v = acc[c][i] + bv;
            v = fmaxf(v, 0.0f);
            Y[(size_t)r * Cout + col] = (half_t)v;
        }
    }
}

// ---------------- masked BN: stats + apply ----------------
__global__ void bn_stats(const half_t* __restrict__ Y, const int* __restrict__ valid,
                         float* __restrict__ stats, int rows, int C, int rowsPerBlock) {
    int c = blockIdx.x * blockDim.x + threadIdx.x;
    if (c >= C) return;
    int r0 = blockIdx.y * rowsPerBlock;
    int r1 = r0 + rowsPerBlock; if (r1 > rows) r1 = rows;
    float s = 0.0f, s2 = 0.0f;
    for (int r = r0; r < r1; ++r) {
        if (!valid || valid[r]) {
            float v = (float)Y[(size_t)r * C + c];
            s += v; s2 += v * v;
        }
    }
    atomicAdd(stats + c, s);
    atomicAdd(stats + 1024 + c, s2);
}

__global__ void bn_apply(half_t* __restrict__ Y, const float* __restrict__ stats,
                         const float* __restrict__ g, const float* __restrict__ beta,
                         long long n, int C) {
    long long i = (long long)blockIdx.x * blockDim.x + threadIdx.x;
    if (i >= n) return;
    int c = (int)(i % C);
    float cnt  = stats[2048];
    float mean = stats[c] / cnt;
    float var  = stats[1024 + c] / cnt - mean * mean;
    float v = (float)Y[i];
    Y[i] = (half_t)((v - mean) * rsqrtf(var + 1e-5f) * g[c] + beta[c]);
}

// ---------------- max-scatter over K valid neighbors ----------------
__global__ void max_scatter(const half_t* __restrict__ Y, const int* __restrict__ valid,
                            half_t* __restrict__ H, int nCenters, int C) {
    long long i = (long long)blockIdx.x * blockDim.x + threadIdx.x;  // nCenters*C
    if (i >= (long long)nCenters * C) return;
    int center = (int)(i / C), c = (int)(i % C);
    float m = -3.0e38f;
    for (int k = 0; k < KNB; ++k) {
        int r = center * KNB + k;
        if (valid[r]) m = fmaxf(m, (float)Y[(size_t)r * C + c]);
    }
    H[(size_t)center * C + c] = (half_t)m;
}

// ---------------- global max-pool, final linear, final BN+relu ----------------
__global__ void global_maxpool(const half_t* __restrict__ Y, float* __restrict__ g,
                               int M, int C) {
    int i = blockIdx.x * blockDim.x + threadIdx.x;  // B*C
    if (i >= B_ * C) return;
    int b = i / C, c = i % C;
    float m = -3.0e38f;
    for (int r = 0; r < M; ++r)
        m = fmaxf(m, (float)Y[((size_t)b * M + r) * C + c]);
    g[i] = m;
}

__global__ void final_linear(const float* __restrict__ g, const float* __restrict__ W,
                             const float* __restrict__ bias, float* __restrict__ tmp) {
    int i = blockIdx.x * blockDim.x + threadIdx.x;  // 8*256
    if (i >= B_ * 256) return;
    int b = i >> 8, c = i & 255;
    float acc = bias[c];
    for (int k = 0; k < 1024; ++k)
        acc += g[b * 1024 + k] * W[(size_t)k * 256 + c];
    tmp[i] = acc;
}

__global__ void final_bn(const float* __restrict__ tmp, const float* __restrict__ g,
                         const float* __restrict__ beta, float* __restrict__ out) {
    int c = threadIdx.x;  // 256 threads
    float mean = 0.0f;
    for (int b = 0; b < B_; ++b) mean += tmp[b * 256 + c];
    mean *= (1.0f / B_);
    float var = 0.0f;
    for (int b = 0; b < B_; ++b) {
        float d = tmp[b * 256 + c] - mean;
        var += d * d;
    }
    var *= (1.0f / B_);
    float inv = rsqrtf(var + 1e-5f);
    for (int b = 0; b < B_; ++b) {
        float v = (tmp[b * 256 + c] - mean) * inv * g[c] + beta[c];
        out[b * 256 + c] = fmaxf(v, 0.0f);
    }
}

// ---------------- host orchestration ----------------
extern "C" void kernel_launch(void* const* d_in, const int* in_sizes, int n_in,
                              void* d_out, int out_size, void* d_ws, size_t ws_size,
                              hipStream_t stream) {
    (void)in_sizes; (void)n_in; (void)out_size; (void)ws_size;
    char* ws = (char*)d_ws;
    const float* data = (const float*)d_in[0];

    float*  pos   = (float*)(ws + O_POS);
    int*    idx1  = (int*)  (ws + O_IDX1);
    float*  cent1 = (float*)(ws + O_CENT1);
    int*    nbr1  = (int*)  (ws + O_NBR1);
    int*    val1  = (int*)  (ws + O_VAL1);
    int*    idx2  = (int*)  (ws + O_IDX2);
    float*  cent2 = (float*)(ws + O_CENT2);
    int*    nbr2  = (int*)  (ws + O_NBR2);
    int*    val2  = (int*)  (ws + O_VAL2);
    half_t* h1    = (half_t*)(ws + O_H1);
    half_t* h2    = (half_t*)(ws + O_H2);
    float*  gbuf  = (float*)(ws + O_G);
    float*  tmpv  = (float*)(ws + O_TMP);
    float*  stats = (float*)(ws + O_STATS);
    half_t* wh    = (half_t*)(ws + O_WH);
    half_t* buf0  = (half_t*)(ws + O_BUF0);
    half_t* buf1  = (half_t*)(ws + O_BUF1);

    // per-layer dims (Kin, Kpad, Cout) and d_in index of W (then b,g,beta follow)
    const int kin[9]  = {3, 64, 64,   131, 128, 128,   259, 256, 512};
    const int kpad[9] = {32, 64, 64,  160, 128, 128,   288, 256, 512};
    const int cout[9] = {64, 64, 128, 128, 128, 256,   256, 512, 1024};
    const int widx[9] = {1, 5, 9,     13, 17, 21,      25, 29, 33};
    size_t woff[9]; size_t wacc = 0;
    for (int i = 0; i < 9; ++i) { woff[i] = wacc; wacc += (size_t)kpad[i] * cout[i]; }

    // 0) transpose input, pack weights into WMMA-fragment layout
    { int n = B_ * N1 * 3; transpose_pos<<<(n + 255) / 256, 256, 0, stream>>>(data, pos, N1); }
    for (int i = 0; i < 9; ++i) {
        int n = kpad[i] * cout[i];
        pack_w<<<(n + 255) / 256, 256, 0, stream>>>(
            (const float*)d_in[widx[i]], wh + woff[i], kin[i], kpad[i], cout[i]);
    }

    auto layer = [&](const half_t* X, half_t* Y, int li, int rows, const int* valid) {
        int Kp = kpad[li], C = cout[li];
        const float* bias = (const float*)d_in[widx[li] + 1];
        const float* gg   = (const float*)d_in[widx[li] + 2];
        const float* bb   = (const float*)d_in[widx[li] + 3];
        long long totalWaves = (long long)(rows / 16) * (C / 64);
        gemm_bias_relu<<<(unsigned)((totalWaves + 7) / 8), 256, 0, stream>>>(
            X, wh + woff[li], bias, Y, rows, Kp, C);
        zero_kernel<<<(2048 + 255) / 256, 256, 0, stream>>>(stats, 2048);
        dim3 sg((unsigned)((C + 255) / 256), (unsigned)((rows + 255) / 256));
        bn_stats<<<sg, 256, 0, stream>>>(Y, valid, stats, rows, C, 256);
        long long n = (long long)rows * C;
        bn_apply<<<(unsigned)((n + 255) / 256), 256, 0, stream>>>(Y, stats, gg, bb, n, C);
    };

    // ---- SA1 ----
    fps_kernel<<<B_, 256, 0, stream>>>(pos, idx1, N1, M1);
    { int n = B_ * M1 * 3; gather_centers<<<(n + 255) / 256, 256, 0, stream>>>(pos, idx1, cent1, N1, M1); }
    ball_query_kernel<<<B_ * M1, 64, 0, stream>>>(pos, cent1, nbr1, val1, N1, M1, 0.04f, KNB);
    set_scalar<<<1, 1, 0, stream>>>(stats + 2048, 0.0f);
    count_valid<<<1024, 256, 0, stream>>>(val1, stats + 2048, E1);
    { long long n = (long long)E1 * 32;
      build_edges1<<<(unsigned)((n + 255) / 256), 256, 0, stream>>>(pos, cent1, nbr1, buf0, N1, M1); }
    layer(buf0, buf1, 0, E1, val1);
    layer(buf1, buf0, 1, E1, val1);
    layer(buf0, buf1, 2, E1, val1);
    { long long n = (long long)B_ * M1 * 128;
      max_scatter<<<(unsigned)((n + 255) / 256), 256, 0, stream>>>(buf1, val1, h1, B_ * M1, 128); }

    // ---- SA2 ----
    fps_kernel<<<B_, 256, 0, stream>>>(cent1, idx2, M1, M2);
    { int n = B_ * M2 * 3; gather_centers<<<(n + 255) / 256, 256, 0, stream>>>(cent1, idx2, cent2, M1, M2); }
    ball_query_kernel<<<B_ * M2, 64, 0, stream>>>(cent1, cent2, nbr2, val2, M1, M2, 0.16f, KNB);
    set_scalar<<<1, 1, 0, stream>>>(stats + 2048, 0.0f);
    count_valid<<<1024, 256, 0, stream>>>(val2, stats + 2048, E2);
    { long long n = (long long)E2 * 160;
      build_edges2<<<(unsigned)((n + 255) / 256), 256, 0, stream>>>(cent1, cent2, nbr2, h1, buf0, M1, M2); }
    layer(buf0, buf1, 3, E2, val2);
    layer(buf1, buf0, 4, E2, val2);
    layer(buf0, buf1, 5, E2, val2);
    { long long n = (long long)B_ * M2 * 256;
      max_scatter<<<(unsigned)((n + 255) / 256), 256, 0, stream>>>(buf1, val2, h2, B_ * M2, 256); }

    // ---- mlp3 + head ----
    { int n = R3 * 288; build_x3<<<(n + 255) / 256, 256, 0, stream>>>(h2, cent2, buf0); }
    set_scalar<<<1, 1, 0, stream>>>(stats + 2048, (float)R3);
    layer(buf0, buf1, 6, R3, nullptr);
    layer(buf1, buf0, 7, R3, nullptr);
    layer(buf0, buf1, 8, R3, nullptr);
    { int n = B_ * 1024; global_maxpool<<<(n + 255) / 256, 256, 0, stream>>>(buf1, gbuf, M2, 1024); }
    { int n = B_ * 256;  final_linear<<<(n + 255) / 256, 256, 0, stream>>>(
          gbuf, (const float*)d_in[37], (const float*)d_in[38], tmpv); }
    final_bn<<<1, 256, 0, stream>>>(tmpv, (const float*)d_in[39], (const float*)d_in[40], (float*)d_out);
}
